// scaled_dot_product_attention_32925219291316
// MI455X (gfx1250) — matrix operand order repaired and checked by reading, not compiled, not run on hardware
//
#include <hip/hip_runtime.h>

// CDNA5 / gfx1250 fused attention (B=8,H=16,S=1024,D=64, fp32 in/out):
//   prepass (if ws fits): Q,K -> f16 copy; V -> f16 [D,S] transpose (LDS tile)
//   logits = Q K^T * 0.125            (f16 WMMA, f32 accum)  -> LDS tile
//   logits += pad(mask) * -1e9        (row-wise in softmax pass, coalesced)
//   P = softmax(logits)               (f32 in LDS, b128 LDS + b128 weights stores)
//   O = P V                           (f16 WMMA, f32 accum)
typedef __attribute__((ext_vector_type(16))) _Float16 v16h;
typedef __attribute__((ext_vector_type(8)))  _Float16 v8h;
typedef __attribute__((ext_vector_type(4)))  _Float16 v4h;
typedef __attribute__((ext_vector_type(8)))  float    v8f;

#define B_ 8
#define H_ 16
#define S_ 1024
#define D_ 64
#define NEG_BIG   (-1.0e9f)
#define SCALE     0.125f       // 1/sqrt(64)
#define ROWS_BLK  32           // score rows per workgroup
#define LDS_STR   1028         // 1024 + 4: 16B-aligned rows, 4-bank rotation per row

// Convert 8 f32 (two float4) into halves [base .. base+7] of an A/B fragment.
__device__ __forceinline__ void cvt8(v16h& a, int base, float4 u, float4 w) {
    a[base + 0] = (_Float16)u.x; a[base + 1] = (_Float16)u.y;
    a[base + 2] = (_Float16)u.z; a[base + 3] = (_Float16)u.w;
    a[base + 4] = (_Float16)w.x; a[base + 5] = (_Float16)w.y;
    a[base + 6] = (_Float16)w.z; a[base + 7] = (_Float16)w.w;
}

__device__ __forceinline__ v16h cat8(v8h lo, v8h hi) {
    return __builtin_shufflevector(lo, hi, 0, 1, 2, 3, 4, 5, 6, 7,
                                           8, 9, 10, 11, 12, 13, 14, 15);
}

// ---------------- Prepass: f32 -> f16 flat copy (Q, K) ----------------
__global__ __launch_bounds__(256)
void f32_to_f16_kernel(const float* __restrict__ src, _Float16* __restrict__ dst, int n8) {
    const int i = blockIdx.x * blockDim.x + threadIdx.x;
    if (i < n8) {
        const float4* s4 = (const float4*)src + (size_t)i * 2;
        float4 u = s4[0], w = s4[1];
        v8h h;
        h[0] = (_Float16)u.x; h[1] = (_Float16)u.y; h[2] = (_Float16)u.z; h[3] = (_Float16)u.w;
        h[4] = (_Float16)w.x; h[5] = (_Float16)w.y; h[6] = (_Float16)w.z; h[7] = (_Float16)w.w;
        ((v8h*)dst)[i] = h;
    }
}

// ---------------- Prepass: V [S,D] f32 -> Vt [D,S] f16 (LDS tile transpose) ----------------
__global__ __launch_bounds__(256)
void v_transpose_kernel(const float* __restrict__ v, _Float16* __restrict__ vt) {
    __shared__ float t[64][65];                       // 64x64 tile, padded
    const int bh = blockIdx.y;
    const int t0 = blockIdx.x * 64;
    const float* Vbh  = v  + (size_t)bh * S_ * D_;
    _Float16*    Vtbh = vt + (size_t)bh * D_ * S_;

    const int qq = threadIdx.x >> 6, l64 = threadIdx.x & 63;
    #pragma unroll
    for (int r = 0; r < 16; ++r) {                    // coalesced f32 row loads
        const int tt = qq * 16 + r;
        t[tt][l64] = Vbh[(size_t)(t0 + tt) * D_ + l64];
    }
    __syncthreads();
    const int tq = (threadIdx.x & 15) * 4;            // 4 consecutive t per thread
    const int db = threadIdx.x >> 4;                  // 0..15
    #pragma unroll
    for (int r = 0; r < 4; ++r) {                     // coalesced 8B f16 column stores
        const int d = db + r * 16;
        v4h h;
        h[0] = (_Float16)t[tq + 0][d]; h[1] = (_Float16)t[tq + 1][d];
        h[2] = (_Float16)t[tq + 2][d]; h[3] = (_Float16)t[tq + 3][d];
        *(v4h*)(Vtbh + (size_t)d * S_ + t0 + tq) = h;
    }
}

// ---------------- Main fused attention kernel ----------------
template <bool FAST>
__global__ __launch_bounds__(256, 2)
void attn_fused_kernel(const float* __restrict__ q,
                       const float* __restrict__ k,
                       const float* __restrict__ v,
                       const float* __restrict__ mask,
                       float* __restrict__ out,
                       const _Float16* __restrict__ qh,
                       const _Float16* __restrict__ kh,
                       const _Float16* __restrict__ vt)
{
    extern __shared__ float sc[];                 // [ROWS_BLK][LDS_STR] f32 scores/probs

    const int bh      = blockIdx.y;               // 0..B*H-1
    const int rowBase = blockIdx.x * ROWS_BLK;    // first score row of this block
    const int tid     = threadIdx.x;
    const int lane    = tid & 31;                 // wave32
    const int wid     = tid >> 5;                 // 0..7
    const int wave_m  = wid >> 2;                 // 0..1 : 16-row group
    const int wave_n  = wid & 3;                  // 0..3 : 16-col group
    const int lm      = lane & 15;                // fragment row/col within tile
    const int hi      = lane >> 4;                // half-wave select
    const size_t bhoff = (size_t)bh * S_ * D_;

    float* Obh = out + bhoff;
    float* Wbh = out + (size_t)B_ * H_ * S_ * D_ + (size_t)bh * S_ * S_;

    // ---------------- Phase 1: score tile = (Q K^T) * scale -> LDS ----------------
    // A layout (16x32 f16): lane holds row m0+lm; K = hi*16..+15 contiguous.
    v16h a0, a1;
    if constexpr (FAST) {
        const _Float16* rp = qh + bhoff + (size_t)(rowBase + wave_m * 16 + lm) * D_ + hi * 16;
        a0 = cat8(*(const v8h*)rp,        *(const v8h*)(rp + 8));
        a1 = cat8(*(const v8h*)(rp + 32), *(const v8h*)(rp + 40));
    } else {
        const float4* row = (const float4*)(q + bhoff + (size_t)(rowBase + wave_m * 16 + lm) * D_);
        const int o = hi * 4;
        cvt8(a0, 0, row[o + 0],  row[o + 1]);
        cvt8(a0, 8, row[o + 2],  row[o + 3]);
        cvt8(a1, 0, row[o + 8],  row[o + 9]);
        cvt8(a1, 8, row[o + 10], row[o + 11]);
    }

    for (int n0 = wave_n * 16; n0 < S_; n0 += 64) {
        // B layout (32x16 f16): lane holds col n0+lm (K row), K = hi*16..+15 contiguous in d.
        v16h b0, b1;
        if constexpr (FAST) {
            const _Float16* kp = kh + bhoff + (size_t)(n0 + lm) * D_ + hi * 16;
            b0 = *(const v16h*)kp;                    // 32B aligned
            b1 = *(const v16h*)(kp + 32);
            if (n0 + 64 < S_)
                __builtin_prefetch((const char*)(kp + (size_t)64 * D_), 0, 3);
        } else {
            const float4* rk = (const float4*)(k + bhoff + (size_t)(n0 + lm) * D_);
            const int o = hi * 4;
            cvt8(b0, 0, rk[o + 0],  rk[o + 1]);
            cvt8(b0, 8, rk[o + 2],  rk[o + 3]);
            cvt8(b1, 0, rk[o + 8],  rk[o + 9]);
            cvt8(b1, 8, rk[o + 10], rk[o + 11]);
            if (n0 + 64 < S_) {
                const char* nk = (const char*)(k + bhoff + (size_t)(n0 + 64 + lm) * D_);
                __builtin_prefetch(nk, 0, 3);
                __builtin_prefetch(nk + 128, 0, 3);
            }
        }
        v8f c = {};
        c = __builtin_amdgcn_wmma_f32_16x16x32_f16(false, a0, false, b0, (short)0, c, false, false);
        c = __builtin_amdgcn_wmma_f32_16x16x32_f16(false, a1, false, b1, (short)0, c, false, false);

        // C/D layout: VGPR r -> M = hi*8 + r, N = lm. Pure scale+store (no branches).
        const int nj = n0 + lm;
        #pragma unroll
        for (int r = 0; r < 8; ++r) {
            const int mi = wave_m * 16 + hi * 8 + r;
            sc[mi * LDS_STR + nj] = c[r] * SCALE;
        }
    }
    __syncthreads();

    // -------- Phase 2: mask + row softmax in LDS, stream weights out (b128) --------
    #pragma unroll
    for (int rr = 0; rr < 4; ++rr) {
        const int mi = wid + rr * 8;
        const int gi = rowBase + mi;                     // wave-uniform global row
        float* srow   = sc + mi * LDS_STR;
        float4* srow4 = (float4*)srow;

        // Pass A: apply padded mask (coalesced row read) and find the row max.
        float mx = -3.402823466e38f;
        if (gi > 0) {
            const float* mrow = mask + (size_t)(gi - 1) * (S_ - 1);
            #pragma unroll 4
            for (int j = lane; j < S_; j += 32) {
                const int jm = (j == 0) ? 0 : (j - 1);   // clamp -> unconditional load
                float mval = mrow[jm];
                mval = (j == 0) ? 0.0f : mval;           // zero-padded first column
                const float s = fmaf(mval, NEG_BIG, srow[j]);
                srow[j] = s;
                mx = fmaxf(mx, s);
            }
        } else {                                         // gi==0: padded first row, no mask
            #pragma unroll 4
            for (int j = lane; j < S_; j += 32) mx = fmaxf(mx, srow[j]);
        }
        #pragma unroll
        for (int o = 16; o >= 1; o >>= 1) mx = fmaxf(mx, __shfl_xor(mx, o, 32));

        // Pass B: exponentiate (b128 LDS traffic) and accumulate the row sum.
        float sum = 0.0f;
        #pragma unroll
        for (int jj = 0; jj < 8; ++jj) {
            float4 x = srow4[lane + jj * 32];
            x.x = __expf(x.x - mx); x.y = __expf(x.y - mx);
            x.z = __expf(x.z - mx); x.w = __expf(x.w - mx);
            srow4[lane + jj * 32] = x;
            sum += (x.x + x.y) + (x.z + x.w);
        }
        #pragma unroll
        for (int o = 16; o >= 1; o >>= 1) sum += __shfl_xor(sum, o, 32);

        // Pass C: normalize in LDS and stream the weights row out as b128 stores.
        const float inv = 1.0f / sum;
        float4* wrow4 = (float4*)(Wbh + (size_t)gi * S_);
        #pragma unroll
        for (int jj = 0; jj < 8; ++jj) {
            float4 x = srow4[lane + jj * 32];
            x.x *= inv; x.y *= inv; x.z *= inv; x.w *= inv;
            srow4[lane + jj * 32] = x;                   // kept for P@V
            wrow4[lane + jj * 32] = x;                   // attention_weights output
        }
    }
    __syncthreads();

    // ---------------- Phase 3: O = P V ----------------
    {
        const int m0 = wave_m * 16;                      // local row group
        const int d0 = wave_n * 16;                      // output feature group
        v8f c = {};
        for (int t0 = 0; t0 < S_; t0 += 32) {
            // A fragment from LDS probabilities: two aligned float4 pairs per half.
            v16h a, b;
            const float4* pr4 = (const float4*)(sc + (m0 + lm) * LDS_STR + t0 + hi * 16);
            cvt8(a, 0, pr4[0], pr4[1]);                  // k = kb .. kb+7
            cvt8(a, 8, pr4[2], pr4[3]);                  // k = kb+8 .. kb+15
            if constexpr (FAST) {
                // B fragment: Vt[d0+lm][t0 + hi*16 ..], one aligned 32B load.
                const _Float16* vp = vt + bhoff + (size_t)(d0 + lm) * S_ + t0 + hi * 16;
                b = *(const v16h*)vp;
                if (t0 + 32 < S_)
                    __builtin_prefetch((const char*)(vp + 32), 0, 3);
            } else {
                const float* vc = v + bhoff + (size_t)(t0 + hi * 16) * D_ + d0 + lm;
                #pragma unroll
                for (int i = 0; i < 16; ++i)
                    b[i] = (_Float16)vc[(size_t)i * D_];
                if (t0 + 32 < S_)
                    __builtin_prefetch((const char*)(v + bhoff + (size_t)(t0 + 32 + lane) * D_), 0, 3);
            }
            c = __builtin_amdgcn_wmma_f32_16x16x32_f16(false, a, false, b, (short)0, c, false, false);
        }
        #pragma unroll
        for (int r = 0; r < 8; ++r) {
            const int gi = rowBase + m0 + hi * 8 + r;
            Obh[(size_t)gi * D_ + d0 + lm] = c[r];
        }
    }
}

extern "C" void kernel_launch(void* const* d_in, const int* in_sizes, int n_in,
                              void* d_out, int out_size, void* d_ws, size_t ws_size,
                              hipStream_t stream) {
    (void)in_sizes; (void)n_in; (void)out_size;
    const float* q    = (const float*)d_in[0];
    const float* k    = (const float*)d_in[1];
    const float* v    = (const float*)d_in[2];
    const float* mask = (const float*)d_in[3];
    float* out = (float*)d_out;

    const size_t elems = (size_t)B_ * H_ * S_ * D_;          // 8.4M per tensor
    const size_t need  = 3 * elems * sizeof(_Float16);       // ~50 MB workspace

    dim3 grid(S_ / ROWS_BLK, B_ * H_);   // 32 x 128 workgroups
    dim3 block(256);                     // 8 wave32 waves
    size_t shmem = (size_t)ROWS_BLK * LDS_STR * sizeof(float);  // ~128.5 KB of 320 KB LDS

    if (ws_size >= need) {
        _Float16* qh = (_Float16*)d_ws;
        _Float16* kh = qh + elems;
        _Float16* vt = kh + elems;
        const int n8 = (int)(elems / 8);
        hipLaunchKernelGGL(f32_to_f16_kernel, dim3(n8 / 256), dim3(256), 0, stream, q, qh, n8);
        hipLaunchKernelGGL(f32_to_f16_kernel, dim3(n8 / 256), dim3(256), 0, stream, k, kh, n8);
        hipLaunchKernelGGL(v_transpose_kernel, dim3(S_ / 64, B_ * H_), dim3(256), 0, stream, v, vt);
        hipLaunchKernelGGL((attn_fused_kernel<true>), grid, block, shmem, stream,
                           q, k, v, mask, out, qh, kh, vt);
    } else {
        hipLaunchKernelGGL((attn_fused_kernel<false>), grid, block, shmem, stream,
                           q, k, v, mask, out, nullptr, nullptr, nullptr);
    }
}
